// SteppingLSTM_86002425135376
// MI455X (gfx1250) — matrix operand order repaired and checked by reading, not compiled, not run on hardware
//
#include <hip/hip_runtime.h>

// ---------------------------------------------------------------------------
// Persistent batch-parallel 2-layer LSTM for MI455X (gfx1250, wave32, WMMA).
//   layer kernel: 16 blocks x 512 threads; block b owns batch rows [16b,16b+16)
//   and runs the whole T-step recurrence with h in LDS (double buffered),
//   c + accumulators in registers, weights pre-packed to bf16 WMMA B-fragments.
//   Weight fragments are re-loaded from (L2-resident) global memory every step
//   through an explicit addrspace(1) pointer whose *uniform* integer base is
//   made opaque per iteration (stops LLVM hoisting them into VGPRs / spilling,
//   keeps them as global_load_b128 instead of flat). The divergent lane offset
//   is added after the asm so the constrained value stays in SGPRs.
//   Gate nonlinearities use the gfx1250 hardware V_TANH_F32 transcendental.
// ---------------------------------------------------------------------------

typedef __bf16 bf16;
typedef __attribute__((ext_vector_type(16))) __bf16 v16bf;
typedef __attribute__((ext_vector_type(8)))  float  v8f;
typedef __attribute__((ext_vector_type(4)))  unsigned int v4u;

typedef const __attribute__((address_space(1))) v16bf* gvec_p;

#define Bsz 256
#define Tn  512
#define Hn  256
#define Gn  1024   // 4*H

__device__ __forceinline__ float fast_tanh(float x) {
#if __has_builtin(__builtin_amdgcn_tanhf)
    return __builtin_amdgcn_tanhf(x);      // V_TANH_F32 (gfx1250 trans op)
#else
    return tanhf(x);
#endif
}
__device__ __forceinline__ float fast_sigmoid(float x) {
    return 0.5f + 0.5f * fast_tanh(0.5f * x);
}

// ---------------------------------------------------------------------------
// Pack fp32 weights into bf16 WMMA-B fragments.
// Fragment (nt, kt) covers B[K = kt*32 .. +31][N = nt*16 .. +15], stored as
// 512 bf16, lane-major (16 contiguous bf16 per lane):
//   lane l < 16 : N = nt*16 + l,      K = kt*32 + q        (q = 0..15)
//   lane l >= 16: N = nt*16 + (l-16), K = kt*32 + 16 + q
// fragment index = nt*KT + kt.
// ---------------------------------------------------------------------------
__global__ void pack_weights_kernel(const float* __restrict__ U0,
                                    const float* __restrict__ W1,
                                    const float* __restrict__ U1,
                                    bf16* __restrict__ pU0,
                                    bf16* __restrict__ pB1)
{
    int tid = blockIdx.x * blockDim.x + threadIdx.x;
    const int N0 = Hn * Gn;        // 262144 elements (U0)
    const int N1 = 2 * Hn * Gn;    // 524288 elements ([W1;U1])
    if (tid < N0) {
        const int KT = Hn / 32;                 // 8
        int e = tid, frag = e >> 9, r = e & 511;
        int lane = r >> 4, q = r & 15;
        int kt = frag % KT, nt = frag / KT;
        int K = kt * 32 + ((lane & 16) ? 16 : 0) + q;
        int N = nt * 16 + (lane & 15);
        pU0[e] = (bf16)U0[K * Gn + N];
    } else if (tid < N0 + N1) {
        const int KT = (2 * Hn) / 32;           // 16
        int e = tid - N0, frag = e >> 9, r = e & 511;
        int lane = r >> 4, q = r & 15;
        int kt = frag % KT, nt = frag / KT;
        int K = kt * 32 + ((lane & 16) ? 16 : 0) + q;
        int N = nt * 16 + (lane & 15);
        float v = (K < Hn) ? W1[K * Gn + N] : U1[(K - Hn) * Gn + N];
        pB1[e] = (bf16)v;
    }
}

// ---------------------------------------------------------------------------
// One LSTM layer, persistent per batch-tile.
//   LAYER==0: A = h (KA=256), input term x[b,t]*W0 added scalar-wise,
//             output y0 written as bf16 to workspace.
//   LAYER==1: A = [y0_t | h] (KA=512), output y1 written fp32 to d_out.
// 512 threads = 16 waves; wave w owns N-tiles nt = w + 16*gate (gate=i,f,g,o),
// so each output column's i/f/g/o live in this wave's 4 accumulators.
// ---------------------------------------------------------------------------
template <int LAYER>
__global__ __launch_bounds__(512)
void lstm_layer_kernel(const bf16*  __restrict__ packedB,  // [NT*KT][512] bf16
                       const float* __restrict__ bias,     // [1024]
                       const float* __restrict__ W0row,    // layer0: [1024]
                       const float* __restrict__ x,        // layer0: [B][T]
                       const bf16*  __restrict__ y0in,     // layer1: [B][T][H] bf16
                       bf16*        __restrict__ y0out,    // layer0 output
                       float*       __restrict__ y1out)    // layer1 output
{
    constexpr int KA   = (LAYER == 0) ? Hn : 2 * Hn;
    constexpr int KT   = KA / 32;
    constexpr int HOFF = KA - Hn;          // h region offset inside A buffer
    constexpr int ROWP = KA + 8;           // +16B row pad -> conflict-free LDS

    __shared__ bf16 Abuf[2][16][ROWP];

    const int tid  = threadIdx.x;
    const int lane = tid & 31;
    const int wv   = tid >> 5;             // wave 0..15
    const int l16  = lane & 15;
    const int hi   = lane >> 4;            // 0: rows 0..7, 1: rows 8..15 (C layout)
    const int b0   = blockIdx.x * 16;

    // Per-lane gate biases / layer-0 input weights for column n = wv*16 + l16.
    const int nb = wv * 16 + l16;
    const float bI = bias[nb],          bF = bias[nb + Hn];
    const float bG = bias[nb + 2 * Hn], bO = bias[nb + 3 * Hn];
    float wI = 0.f, wF = 0.f, wG = 0.f, wO = 0.f;
    if constexpr (LAYER == 0) {
        wI = W0row[nb];          wF = W0row[nb + Hn];
        wG = W0row[nb + 2 * Hn]; wO = W0row[nb + 3 * Hn];
    }

    // Init buffer 0: h = 0; (layer1) y0 tile for t = 0.
    for (int e = tid; e < 16 * Hn; e += 512) {
        Abuf[0][e >> 8][HOFF + (e & 255)] = (bf16)0.0f;
    }
    if constexpr (LAYER == 1) {
        int m = tid >> 5, col = (tid & 31) * 8;          // 512 chunks of 8 bf16
        const v4u* src = reinterpret_cast<const v4u*>(
            &y0in[((size_t)(b0 + m) * Tn + 0) * Hn + col]);
        *reinterpret_cast<v4u*>(&Abuf[0][m][col]) = *src;
    }
    __syncthreads();

    float cst[8];
    #pragma unroll
    for (int r = 0; r < 8; ++r) cst[r] = 0.0f;

    // Per-lane byte offset of this lane's 32B slice inside a fragment
    // (divergent -> must NOT be folded into the SGPR-constrained base).
    const unsigned long long laneByte = (unsigned long long)lane * 32ull;

    for (int t = 0; t < Tn; ++t) {
        const int cur = t & 1, nxt = cur ^ 1;

        // Opaque UNIFORM 64-bit base: redefined every iteration so the
        // fragment loads are not loop-invariant (no hoist/spill); the
        // addrspace(1) cast keeps them as global_load_b128 (LOADcnt only).
        unsigned long long pbase = (unsigned long long)(uintptr_t)packedB;
        asm volatile("" : "+s"(pbase));

        v8f acc[4] = {};                    // i, f, g, o accumulators

        #pragma unroll 2
        for (int kt = 0; kt < KT; ++kt) {
            // A fragment: lane<16 -> row l16, K = kt*32 + {0..15};
            //             lane>=16 -> same row, K = kt*32 + {16..31} (matches B packing).
            union { v16bf v; v4u u[2]; } af;
            const bf16* ap = &Abuf[cur][l16][kt * 32 + hi * 16];
            af.u[0] = *reinterpret_cast<const v4u*>(ap);
            af.u[1] = *reinterpret_cast<const v4u*>(ap + 8);

            // Load all 4 gate fragments first (clauseable b128 global loads),
            // then run the 4 WMMAs back-to-back.
            v16bf bfv[4];
            #pragma unroll
            for (int g = 0; g < 4; ++g) {
                const unsigned long long frag =
                    (unsigned long long)((wv + 16 * g) * KT + kt);
                bfv[g] = *(gvec_p)(pbase + frag * 1024ull + laneByte);
            }
            #pragma unroll
            for (int g = 0; g < 4; ++g) {
                acc[g] = __builtin_amdgcn_wmma_f32_16x16x32_bf16(
                    false, af.v, false, bfv[g], (short)0, acc[g], false, false);
            }
        }

        // Layer-0 scalar input contribution x[b,t] * W0 (D = 1).
        float xv[8];
        if constexpr (LAYER == 0) {
            #pragma unroll
            for (int r = 0; r < 8; ++r)
                xv[r] = x[(size_t)(b0 + hi * 8 + r) * Tn + t];
        }

        // Gates + state update. acc[g][r] is z for row m = hi*8 + r, col nb.
        #pragma unroll
        for (int r = 0; r < 8; ++r) {
            float zi = acc[0][r] + bI;
            float zf = acc[1][r] + bF;
            float zg = acc[2][r] + bG;
            float zo = acc[3][r] + bO;
            if constexpr (LAYER == 0) {
                zi += xv[r] * wI; zf += xv[r] * wF;
                zg += xv[r] * wG; zo += xv[r] * wO;
            }
            float ig = fast_sigmoid(zi);
            float fg = fast_sigmoid(zf);
            float gg = fast_tanh(zg);
            float og = fast_sigmoid(zo);
            float cn = fg * cst[r] + ig * gg;
            cst[r] = cn;
            float hn = og * fast_tanh(cn);

            const int m = hi * 8 + r;
            Abuf[nxt][m][HOFF + nb] = (bf16)hn;     // h for next step
            const size_t oidx = ((size_t)(b0 + m) * Tn + t) * Hn + nb;
            if constexpr (LAYER == 0) y0out[oidx] = (bf16)hn;
            else                      y1out[oidx] = hn;
        }

        // Layer 1: stage next step's y0 tile into the other buffer.
        if constexpr (LAYER == 1) {
            if (t + 1 < Tn) {
                int m = tid >> 5, col = (tid & 31) * 8;
                const v4u* src = reinterpret_cast<const v4u*>(
                    &y0in[((size_t)(b0 + m) * Tn + (t + 1)) * Hn + col]);
                *reinterpret_cast<v4u*>(&Abuf[nxt][m][col]) = *src;
            }
        }
        __syncthreads();
    }
}

// ---------------------------------------------------------------------------
// Launch: pack weights once, then layer 0 then layer 1 (both persistent).
// Workspace layout: [0, 512K) packed U0 | [512K, 1.5M) packed [W1;U1]
//                   | [1.5M, 1.5M + 64M) y0 in bf16.
// ---------------------------------------------------------------------------
extern "C" void kernel_launch(void* const* d_in, const int* in_sizes, int n_in,
                              void* d_out, int out_size, void* d_ws, size_t ws_size,
                              hipStream_t stream)
{
    (void)in_sizes; (void)n_in; (void)out_size; (void)ws_size;
    const float* x  = (const float*)d_in[0];
    const float* W0 = (const float*)d_in[1];
    const float* U0 = (const float*)d_in[2];
    const float* b0 = (const float*)d_in[3];
    const float* W1 = (const float*)d_in[4];
    const float* U1 = (const float*)d_in[5];
    const float* b1 = (const float*)d_in[6];
    float* out = (float*)d_out;

    char* ws  = (char*)d_ws;
    bf16* pU0 = (bf16*)(ws);                          // 512 KB
    bf16* pB1 = (bf16*)(ws + (512ull << 10));         // 1 MB
    bf16* y0  = (bf16*)(ws + (1536ull << 10));        // 64 MB

    const int totalPack = Hn * Gn + 2 * Hn * Gn;      // 786432
    pack_weights_kernel<<<(totalPack + 255) / 256, 256, 0, stream>>>(
        U0, W1, U1, pU0, pB1);

    lstm_layer_kernel<0><<<dim3(Bsz / 16), dim3(512), 0, stream>>>(
        pU0, b0, W0, x, (const bf16*)nullptr, y0, (float*)nullptr);

    lstm_layer_kernel<1><<<dim3(Bsz / 16), dim3(512), 0, stream>>>(
        pB1, b1, (const float*)nullptr, (const float*)nullptr, y0,
        (bf16*)nullptr, out);
}
